// Capsule_62414464746053
// MI455X (gfx1250) — matrix operand order repaired and checked by reading, not compiled, not run on hardware
//
#include <hip/hip_runtime.h>
#include <hip/hip_bf16.h>
#include <math.h>

// ---------------------------------------------------------------------------
// Capsule dynamic routing for MI455X (gfx1250, wave32, WMMA)
//   u_vecs: (64, 2048, 128) fp32     W: (128, 160) fp32
//   u_hat = u_vecs @ W  -> stored f16 in workspace (halves routing traffic)
//   3 routing iterations (b is *replaced* each iter in this reference),
//   final squash -> (64, 10, 16) fp32
// ---------------------------------------------------------------------------

typedef __attribute__((ext_vector_type(16))) _Float16 v16h;
typedef __attribute__((ext_vector_type(8)))  _Float16 h8;
typedef __attribute__((ext_vector_type(4)))  _Float16 h4;
typedef __attribute__((ext_vector_type(2)))  _Float16 h2;
typedef __attribute__((ext_vector_type(8)))  float    v8f;

#define B_BATCH   64
#define N_IN      2048
#define K_IN      128
#define NCAP      10
#define DCAP      16
#define NCOLS     160              // NCAP * DCAP
#define M_TOTAL   (B_BATCH * N_IN) // 131072
#define M_BLK     128
#define KC        32               // K-chunk per LDS stage
#define LDS_STRIDE 40              // 32 halves + 8 pad (80B rows, 16B aligned)
#define OUT_STRIDE 88              // 80 halves + 8 pad (176B rows, 16B aligned)

// ---------------------------------------------------------------------------
// Kernel 1: u_hat GEMM with v_wmma_f32_16x16x32_f16.
// Block = 256 threads = 8 waves; each wave computes a 16x160 strip.
// ---------------------------------------------------------------------------
__global__ __launch_bounds__(256, 1)
void gemm_uhat(const float* __restrict__ U, const float* __restrict__ W,
               _Float16* __restrict__ UH) {
    __shared__ __align__(16) _Float16 sA[M_BLK][LDS_STRIDE];   // 128 rows x 32 k
    __shared__ __align__(16) _Float16 sB[NCOLS][LDS_STRIDE];   // 160 cols x 32 k (Wt)
    __shared__ __align__(16) _Float16 sOut[M_BLK][OUT_STRIDE]; // 128 x 80 epilogue stage

    const int tid  = threadIdx.x;
    const int lane = tid & 31;
    const int wave = tid >> 5;
    const long rowBase = (long)blockIdx.x * M_BLK;

    v8f acc[NCAP];
    #pragma unroll
    for (int j = 0; j < NCAP; ++j)
        #pragma unroll
        for (int q = 0; q < 8; ++q) acc[j][q] = 0.0f;

    // A-fragment addressing (ISA 7.12.2, 16-bit A 16x32):
    //   lanes 0-15 : M = lane,    K in {0..7, 16..23}
    //   lanes 16-31: M = lane-16, K in {8..15, 24..31}
    const int rowA   = wave * 16 + (lane & 15);
    const int khalfA = (lane < 16) ? 0 : 8;
    // B-fragment: col = lane%16, K in {0..7,16..23} / {8..15,24..31} (matches A).
    const int colLo  = lane & 15;
    const int kbB    = (lane < 16) ? 0 : 8;

    for (int kc = 0; kc < K_IN / KC; ++kc) {
        // --- stage A: 128 rows x 32 k; float4 -> packed 4-half ds_store_b64 ---
        #pragma unroll
        for (int i = 0; i < 4; ++i) {
            const int li = tid + i * 256;        // 0..1023
            const int r  = li >> 3;              // row 0..127
            const int kq = (li & 7) * 4;         // k 0..28 step 4
            const float4 f = *(const float4*)(U + (rowBase + r) * K_IN + kc * KC + kq);
            h4 hv;
            hv[0] = (_Float16)f.x; hv[1] = (_Float16)f.y;
            hv[2] = (_Float16)f.z; hv[3] = (_Float16)f.w;
            *(h4*)&sA[r][kq] = hv;               // 8B aligned (80B row stride)
        }
        // --- stage W transposed: k-pairs packed into 4B ds_store_b32 ---
        #pragma unroll
        for (int i = 0; i < 10; ++i) {
            const int li  = tid + i * 256;       // 0..2559
            const int kp  = li / NCOLS;          // k-pair 0..15
            const int col = li % NCOLS;          // coalesced global reads
            const float f0 = W[(kc * KC + 2 * kp)     * NCOLS + col];
            const float f1 = W[(kc * KC + 2 * kp + 1) * NCOLS + col];
            h2 hv; hv[0] = (_Float16)f0; hv[1] = (_Float16)f1;
            *(h2*)&sB[col][2 * kp] = hv;         // 4B aligned
        }
        __syncthreads();

        // A fragment: two aligned 16B LDS loads -> v16h
        v16h a;
        {
            const h8 alo = *(const h8*)&sA[rowA][khalfA];
            const h8 ahi = *(const h8*)&sA[rowA][16 + khalfA];
            #pragma unroll
            for (int q = 0; q < 8; ++q) { a[q] = alo[q]; a[8 + q] = ahi[q]; }
        }

        #pragma unroll
        for (int jt = 0; jt < NCAP; ++jt) {
            const int col = jt * 16 + colLo;
            v16h bm;
            const h8 blo = *(const h8*)&sB[col][kbB];
            const h8 bhi = *(const h8*)&sB[col][kbB + 16];
            #pragma unroll
            for (int q = 0; q < 8; ++q) { bm[q] = blo[q]; bm[8 + q] = bhi[q]; }
            acc[jt] = __builtin_amdgcn_wmma_f32_16x16x32_f16(
                false, a, false, bm, (short)0, acc[jt], false, false);
        }
        __syncthreads();
    }

    // Epilogue: C/D layout (VGPR v -> M=v for lanes<16, M=v+8 for lanes>=16).
    // Stage through LDS in two 128x80 passes so global stores are b128/lane.
    const int rvoff = (lane < 16) ? 0 : 8;
    #pragma unroll
    for (int p = 0; p < 2; ++p) {
        #pragma unroll
        for (int jt5 = 0; jt5 < 5; ++jt5) {
            const int jt = p * 5 + jt5;
            #pragma unroll
            for (int v = 0; v < 8; ++v)
                sOut[wave * 16 + rvoff + v][jt5 * 16 + colLo] = (_Float16)acc[jt][v];
        }
        __syncthreads();
        // 128 rows x 80 halves = 1280 16B chunks; 5 per thread, full-line stores
        #pragma unroll
        for (int i = 0; i < 5; ++i) {
            const int li = tid + i * 256;        // 0..1279
            const int r  = li / 10;
            const int cc = li % 10;              // 8-half chunk within 80
            const h8 vdat = *(const h8*)&sOut[r][cc * 8];
            *(h8*)(UH + (rowBase + r) * NCOLS + p * 80 + cc * 8) = vdat;
        }
        __syncthreads();
    }
}

// ---------------------------------------------------------------------------
// Kernel: zero a float buffer (atomic targets must start at 0 every call)
// ---------------------------------------------------------------------------
__global__ void zero_f32(float* __restrict__ p, int n) {
    const int i = blockIdx.x * 256 + threadIdx.x;
    if (i < n) p[i] = 0.0f;
}

// ---------------------------------------------------------------------------
// Kernel 2: o0[b,col] = 0.1 * sum_n u_hat[b,n,col]   (softmax(0) = 1/10)
// grid (B, 16) x 160 threads; each block reduces 128 n's, atomicAdd partials.
// ---------------------------------------------------------------------------
__global__ void colsum0(const _Float16* __restrict__ UH, float* __restrict__ o0) {
    const int b   = blockIdx.x;
    const int ch  = blockIdx.y;
    const int col = threadIdx.x;
    const _Float16* p = UH + ((long)b * N_IN + ch * 128) * NCOLS + col;
    float s = 0.0f;
    #pragma unroll 4
    for (int n = 0; n < 128; ++n) s += (float)p[n * NCOLS];
    atomicAdd(&o0[b * NCOLS + col], 0.1f * s);
}

// ---------------------------------------------------------------------------
// Kernel 3a: per (b,n): logit[j] = <o_prev[b,j,:], u_hat[b,n,j,:]>,
//            c[b,j,n] = softmax_j(logit)   (max-subtracted, like jax)
// grid 512 x 256 threads; o_prev (160 f32) cached in LDS.
// ---------------------------------------------------------------------------
__global__ __launch_bounds__(256, 1)
void logits_softmax(const _Float16* __restrict__ UH, const float* __restrict__ oPrev,
                    float* __restrict__ C) {
    __shared__ float so[NCOLS];
    const int b = blockIdx.x >> 3;
    const int n = (blockIdx.x & 7) * 256 + threadIdx.x;
    if (threadIdx.x < NCOLS) so[threadIdx.x] = oPrev[b * NCOLS + threadIdx.x];
    __syncthreads();

    const _Float16* row = UH + ((long)b * N_IN + n) * NCOLS;
    float l[NCAP];
    float mx = -3.4e38f;
    #pragma unroll
    for (int j = 0; j < NCAP; ++j) {
        const h8 u0 = *(const h8*)(row + j * 16);
        const h8 u1 = *(const h8*)(row + j * 16 + 8);
        float s = 0.0f;
        #pragma unroll
        for (int d = 0; d < 8; ++d)
            s += so[j * 16 + d] * (float)u0[d] + so[j * 16 + 8 + d] * (float)u1[d];
        l[j] = s;
        mx = fmaxf(mx, s);
    }
    float sum = 0.0f;
    #pragma unroll
    for (int j = 0; j < NCAP; ++j) { l[j] = __expf(l[j] - mx); sum += l[j]; }
    const float inv = 1.0f / sum;
    #pragma unroll
    for (int j = 0; j < NCAP; ++j)
        C[((long)b * NCAP + j) * N_IN + n] = l[j] * inv;   // coalesced per j
}

// ---------------------------------------------------------------------------
// Kernel 3b: o_next[b,col] = sum_n c[b, col/16, n] * u_hat[b,n,col]
// ---------------------------------------------------------------------------
__global__ void weighted_sum(const _Float16* __restrict__ UH, const float* __restrict__ C,
                             float* __restrict__ oNext) {
    const int b   = blockIdx.x;
    const int ch  = blockIdx.y;
    const int col = threadIdx.x;
    const int j   = col >> 4;
    const _Float16* p  = UH + ((long)b * N_IN + ch * 128) * NCOLS + col;
    const float*    cp = C  + ((long)b * NCAP + j) * N_IN + ch * 128;
    float s = 0.0f;
    #pragma unroll 4
    for (int n = 0; n < 128; ++n) s += cp[n] * (float)p[n * NCOLS];
    atomicAdd(&oNext[b * NCOLS + col], s);
}

// ---------------------------------------------------------------------------
// Kernel 4: squash: y = x * s2 / ((1+s2) * sqrt(s2 + eps)), s2 over D=16.
// 16-lane shuffle reduction (groups of 16 lanes == one capsule vector).
// ---------------------------------------------------------------------------
__global__ void squash_k(const float* __restrict__ o, float* __restrict__ out) {
    const int i = blockIdx.x * 256 + threadIdx.x;   // 0..10239
    const float x = o[i];
    float s2 = x * x;
    #pragma unroll
    for (int off = 1; off < 16; off <<= 1)
        s2 += __shfl_xor(s2, off, 16);
    out[i] = x * s2 / ((1.0f + s2) * sqrtf(s2 + 1e-7f));
}

// ---------------------------------------------------------------------------
extern "C" void kernel_launch(void* const* d_in, const int* in_sizes, int n_in,
                              void* d_out, int out_size, void* d_ws, size_t ws_size,
                              hipStream_t stream) {
    const float* U = (const float*)d_in[0];   // (64, 2048, 128)
    const float* W = (const float*)d_in[1];   // (128, 160)
    float* out = (float*)d_out;               // (64, 10, 16)

    // Workspace layout
    char* ws = (char*)d_ws;
    _Float16* UH = (_Float16*)ws;                                   // 41,943,040 B
    size_t off = (size_t)M_TOTAL * NCOLS * sizeof(_Float16);
    float* C  = (float*)(ws + off);                                 //  5,242,880 B
    off += (size_t)B_BATCH * NCAP * N_IN * sizeof(float);
    float* oA = (float*)(ws + off);                                 //     40,960 B
    off += (size_t)B_BATCH * NCOLS * sizeof(float);
    float* oB = (float*)(ws + off);                                 //     40,960 B

    const int OSZ = B_BATCH * NCOLS;          // 10240

    // 1) u_hat = u_vecs @ W   (WMMA f16 -> f32, stored f16)
    gemm_uhat<<<M_TOTAL / M_BLK, 256, 0, stream>>>(U, W, UH);

    // 2) iter 0: c = 1/10 exactly -> o0 = 0.1 * colsum
    zero_f32<<<(OSZ + 255) / 256, 256, 0, stream>>>(oA, OSZ);
    colsum0<<<dim3(B_BATCH, 16), NCOLS, 0, stream>>>(UH, oA);

    // 3) iter 1: b = <o0, u_hat>; c = softmax; o1 = sum c*u_hat
    logits_softmax<<<B_BATCH * 8, 256, 0, stream>>>(UH, oA, C);
    zero_f32<<<(OSZ + 255) / 256, 256, 0, stream>>>(oB, OSZ);
    weighted_sum<<<dim3(B_BATCH, 16), NCOLS, 0, stream>>>(UH, C, oB);

    // 4) iter 2: b = <o1, u_hat>; c = softmax; o2 = sum c*u_hat
    logits_softmax<<<B_BATCH * 8, 256, 0, stream>>>(UH, oB, C);
    zero_f32<<<(OSZ + 255) / 256, 256, 0, stream>>>(oA, OSZ);
    weighted_sum<<<dim3(B_BATCH, 16), NCOLS, 0, stream>>>(UH, C, oA);

    // 5) squash(o2) -> out
    squash_k<<<(OSZ + 255) / 256, 256, 0, stream>>>(oA, out);
}
